// TGADecoder_18949395710253
// MI455X (gfx1250) — matrix operand order repaired and checked by reading, not compiled, not run on hardware
//
#include <hip/hip_runtime.h>
#include <hip/hip_bf16.h>
#include <math.h>
#include <stdint.h>

// Problem constants (match reference)
#define NN     100000
#define NPOOL  50000
#define EDGES  1600000

typedef __attribute__((ext_vector_type(16))) _Float16 v16h;
typedef __attribute__((ext_vector_type(8)))  _Float16 v8h;
typedef __attribute__((ext_vector_type(8)))  float    v8f;

#define BM 64
#define BN 64
#define BK 32

#define ACT_LIN  0
#define ACT_RELU 1
#define ACT_MUL  2

// gfx1250 async global->LDS staging (GLOBAL_LOAD_ASYNC_TO_LDS_B128, ASYNCcnt)
#if defined(__has_builtin)
#if __has_builtin(__builtin_amdgcn_global_load_async_to_lds_b128) && \
    __has_builtin(__builtin_amdgcn_s_wait_asynccnt)
#define USE_ASYNC_LDS 1
#endif
#endif

#ifdef USE_ASYNC_LDS
// exact parameter types per hipcc diagnostics:
//   param 1: int __attribute__((vector_size(16))) __device__(AS1) *
//   param 2: int __attribute__((vector_size(16))) __shared__(AS3) *
typedef int v4i32 __attribute__((vector_size(4 * sizeof(int))));
typedef __attribute__((address_space(1))) v4i32 as1_v4i32;
typedef __attribute__((address_space(3))) v4i32 as3_v4i32;
#endif

// ---------------- order-preserving float<->uint key for atomic max ----------
__device__ __forceinline__ unsigned fkey(float f) {
    unsigned u = __float_as_uint(f);
    return (u & 0x80000000u) ? ~u : (u | 0x80000000u);
}
__device__ __forceinline__ float funkey(unsigned k) {
    unsigned u = (k & 0x80000000u) ? (k & 0x7FFFFFFFu) : ~k;
    return __uint_as_float(u);
}

// ---------------- weight transpose + f32->f16 convert -----------------------
// WT[n][k] = (half) W[k][n]   (W is K x nout row-major)
__global__ void k_cvt_wt(const float* __restrict__ W, _Float16* __restrict__ WT,
                         int K, int nout) {
    long long t = (long long)blockIdx.x * blockDim.x + threadIdx.x;
    if (t >= (long long)K * nout) return;
    int k = (int)(t / nout), n = (int)(t % nout);
    WT[(size_t)n * K + k] = (_Float16)W[t];
}

// ---------------- elementwise kernels ---------------------------------------
__global__ void k_copy2(const float* __restrict__ s, float* __restrict__ d,
                        _Float16* __restrict__ d16, long long n) {
    long long t = (long long)blockIdx.x * blockDim.x + threadIdx.x;
    if (t < n) { float v = s[t]; d[t] = v; d16[t] = (_Float16)v; }
}
__global__ void k_zero(float* __restrict__ p, long long n) {
    long long t = (long long)blockIdx.x * blockDim.x + threadIdx.x;
    if (t < n) p[t] = 0.f;
}
// h = zeros.at[pool_idx].set(features) + fused  (pool rows overwritten)
__global__ void k_unpool2(const float* __restrict__ feats, const float* __restrict__ fused,
                          const int* __restrict__ pidx, float* __restrict__ H,
                          _Float16* __restrict__ H16) {
    long long t = (long long)blockIdx.x * blockDim.x + threadIdx.x;
    if (t >= (long long)NPOOL * 128) return;
    int i = (int)(t >> 7), c = (int)(t & 127);
    int row = pidx[i];
    float v = feats[t] + fused[(size_t)row * 128 + c];
    H[(size_t)row * 128 + c] = v;
    H16[(size_t)row * 128 + c] = (_Float16)v;
}
// el[n,h] = sum_d feat[n,h,d]*al[h,d] ; one block (128 thr = 4 waves) per node
__global__ void k_el_er(const float* __restrict__ F, const float* __restrict__ al,
                        const float* __restrict__ ar, float* __restrict__ EL,
                        float* __restrict__ ER) {
    int n = blockIdx.x;
    int tid = threadIdx.x;              // h = tid>>5, d = tid&31
    int h = tid >> 5, lane = tid & 31;
    float f  = F[(size_t)n * 128 + tid];
    float pl = f * al[tid];
    float pr = f * ar[tid];
    for (int off = 16; off > 0; off >>= 1) {
        pl += __shfl_down(pl, off, 32);
        pr += __shfl_down(pr, off, 32);
    }
    if (lane == 0) { EL[(size_t)n * 4 + h] = pl; ER[(size_t)n * 4 + h] = pr; }
}
// pass1: e = leaky_relu(el[src]+er[dst]); store; segment max via key atomicMax
__global__ void k_edge1(const int* __restrict__ src, const int* __restrict__ dst,
                        const float* __restrict__ EL, const float* __restrict__ ER,
                        float* __restrict__ ELOG, unsigned* __restrict__ EMAX) {
    long long t = (long long)blockIdx.x * blockDim.x + threadIdx.x;
    if (t >= (long long)EDGES * 4) return;
    int e = (int)(t >> 2), h = (int)(t & 3);
    int s = src[e], d = dst[e];
    float x = EL[(size_t)s * 4 + h] + ER[(size_t)d * 4 + h];
    x = x > 0.f ? x : 0.2f * x;
    ELOG[t] = x;
    atomicMax(&EMAX[(size_t)d * 4 + h], fkey(x));
}
// pass2: ez = exp(e - max[dst]); denom[dst] += ez
__global__ void k_edge2(const int* __restrict__ dst, float* __restrict__ ELOG,
                        const unsigned* __restrict__ EMAX, float* __restrict__ DEN) {
    long long t = (long long)blockIdx.x * blockDim.x + threadIdx.x;
    if (t >= (long long)EDGES * 4) return;
    int e = (int)(t >> 2), h = (int)(t & 3);
    int d = dst[e];
    float m = funkey(EMAX[(size_t)d * 4 + h]);
    float z = expf(ELOG[t] - m);
    ELOG[t] = z;
    atomicAdd(&DEN[(size_t)d * 4 + h], z);
}
// pass3: out[dst, c] += feat[src, c] * alpha[e, h]   (c = h*32+d)
__global__ void k_edge3(const int* __restrict__ src, const int* __restrict__ dst,
                        const float* __restrict__ ELOG, const float* __restrict__ DEN,
                        const float* __restrict__ F, float* __restrict__ OUT, int ldc) {
    long long t = (long long)blockIdx.x * blockDim.x + threadIdx.x;
    if (t >= (long long)EDGES * 128) return;
    int e = (int)(t >> 7), c = (int)(t & 127), h = c >> 5;
    int s = src[e], d = dst[e];
    float alpha = ELOG[(size_t)e * 4 + h] / DEN[(size_t)d * 4 + h];
    atomicAdd(&OUT[(size_t)d * ldc + c], F[(size_t)s * 128 + c] * alpha);
}
// X[:, :128] = elu(X + b) in place (row stride ldc) + f16 shadow
__global__ void k_bias_elu(float* __restrict__ X, int ldc, const float* __restrict__ b,
                           _Float16* __restrict__ X16) {
    long long t = (long long)blockIdx.x * blockDim.x + threadIdx.x;
    if (t >= (long long)NN * 128) return;
    int n = (int)(t >> 7), c = (int)(t & 127);
    float x = X[(size_t)n * ldc + c] + b[c];
    x = x > 0.f ? x : (expf(x) - 1.f);
    X[(size_t)n * ldc + c] = x;
    X16[(size_t)n * ldc + c] = (_Float16)x;
}
// GCT gating: c *= sigmoid(rowmax(c)) + sigmoid(rowmean(c)) ; 128 thr per node
__global__ void k_gate(float* __restrict__ Cm, _Float16* __restrict__ C16) {
    int n = blockIdx.x;
    int tid = threadIdx.x;
    int w = tid >> 5, lane = tid & 31;
    __shared__ float smax[4], ssum[4];
    float v = Cm[(size_t)n * 128 + tid];
    float m = v, s = v;
    for (int off = 16; off > 0; off >>= 1) {
        m = fmaxf(m, __shfl_down(m, off, 32));
        s += __shfl_down(s, off, 32);
    }
    if (lane == 0) { smax[w] = m; ssum[w] = s; }
    __syncthreads();
    if (tid == 0) {
        smax[0] = fmaxf(fmaxf(smax[0], smax[1]), fmaxf(smax[2], smax[3]));
        ssum[0] = ssum[0] + ssum[1] + ssum[2] + ssum[3];
    }
    __syncthreads();
    float g = 1.f / (1.f + expf(-smax[0])) + 1.f / (1.f + expf(-(ssum[0] * (1.f / 128.f))));
    float o = v * g;
    Cm[(size_t)n * 128 + tid] = o;
    C16[(size_t)n * 128 + tid] = (_Float16)o;
}

// ---------------- WMMA GEMM (f16 in, f32 acc) -------------------------------
// C[M, nout] = act( A16[M, Kdim](lda halves) @ WT^T + bias )
// WT is nout x Kdim row-major f16 (pre-transposed weights).
// 256 threads = 8 waves (4 m-groups x 2 n-groups); wave -> 16 rows x 32 cols.
__global__ __launch_bounds__(256)
void k_gemm(const _Float16* __restrict__ A, int lda,
            const _Float16* __restrict__ WT,
            const float* __restrict__ bias,
            float* __restrict__ C, _Float16* __restrict__ C16, int ldc,
            const float* __restrict__ aux,
            int M, int Kdim, int act) {
    // 80-byte row stride: 16B-aligned vector accesses, conflict-friendly
    __shared__ _Float16 As[BM][BK + 8];   // [row][k]
    __shared__ _Float16 Bs[BN][BK + 8];   // [col][k]  (K-contiguous per column)

    const int tid  = threadIdx.x;
    const int lane = tid & 31;
    const int wave = tid >> 5;
    const int wm   = wave >> 1;           // 0..3  row group of 16
    const int wn   = wave & 1;            // 0..1  col group of 32
    const int row0 = blockIdx.x * BM;
    const int col0 = blockIdx.y * BN;

    const int r   = lane & 15;
    const int kh8 = (lane >> 4) << 4;     // A frag: K base 0/16, 16 contiguous
    const int kb  = (lane >> 4) << 4;     // B frag: K base 0/16, 16 contiguous

    // staging: each thread moves one 16B chunk (8 halves) per tile per k-step
    const int srow = tid >> 2;            // 0..63
    const int sq   = (tid & 3) << 3;      // 0,8,16,24 halves

    union { v8f v; float e[8]; } acc0, acc1;
#pragma unroll
    for (int i = 0; i < 8; ++i) { acc0.e[i] = 0.f; acc1.e[i] = 0.f; }

    // Clamp overhang rows to a valid row instead of predicating: the duplicated
    // data only feeds accumulator lanes whose stores are guarded below, and it
    // keeps EXEC fully active through the hot loop (required for async-to-LDS).
    int sgrow = row0 + srow; if (sgrow > M - 1) sgrow = M - 1;
    const _Float16* aRow = A + (size_t)sgrow * lda;
    const _Float16* wRow = WT + (size_t)(col0 + srow) * Kdim;

    for (int k0 = 0; k0 < Kdim; k0 += BK) {
        // prefetch next A k-slice (global_prefetch_b8)
        if (k0 + BK < Kdim)
            __builtin_prefetch(aRow + k0 + BK + sq, 0, 1);

#ifdef USE_ASYNC_LDS
        __builtin_amdgcn_global_load_async_to_lds_b128(
            (as1_v4i32*)(uintptr_t)(aRow + k0 + sq),
            (as3_v4i32*)(uintptr_t)(&As[srow][sq]), 0, 0);
        __builtin_amdgcn_global_load_async_to_lds_b128(
            (as1_v4i32*)(uintptr_t)(wRow + k0 + sq),
            (as3_v4i32*)(uintptr_t)(&Bs[srow][sq]), 0, 0);
        __builtin_amdgcn_s_wait_asynccnt(0);
#else
        *(uint4*)(&As[srow][sq]) = *(const uint4*)(aRow + k0 + sq);
        *(uint4*)(&Bs[srow][sq]) = *(const uint4*)(wRow + k0 + sq);
#endif
        __syncthreads();

        union { v16h v; v8h h[2]; } af, bf0, bf1;
        const _Float16* ap = &As[wm * 16 + r][0];
        af.h[0] = *(const v8h*)(ap + kh8);
        af.h[1] = *(const v8h*)(ap + kh8 + 8);
        const _Float16* bp0 = &Bs[wn * 32 + r][kb];
        bf0.h[0] = *(const v8h*)(bp0);
        bf0.h[1] = *(const v8h*)(bp0 + 8);
        const _Float16* bp1 = &Bs[wn * 32 + 16 + r][kb];
        bf1.h[0] = *(const v8h*)(bp1);
        bf1.h[1] = *(const v8h*)(bp1 + 8);

        acc0.v = __builtin_amdgcn_wmma_f32_16x16x32_f16(false, af.v, false, bf0.v,
                                                        (short)0, acc0.v, false, false);
        acc1.v = __builtin_amdgcn_wmma_f32_16x16x32_f16(false, af.v, false, bf1.v,
                                                        (short)0, acc1.v, false, false);
        __syncthreads();
    }

    // C/D layout: VGPR i, lanes 0-15 -> M=i, lanes 16-31 -> M=8+i ; N = lane&15
    const int rbase = row0 + wm * 16 + ((lane >> 4) << 3);
    const int nc0   = wn * 32 + r;
#pragma unroll
    for (int i = 0; i < 8; ++i) {
        int grow = rbase + i;
        if (grow >= M) continue;
        {
            int wcol = col0 + nc0;
            float x = acc0.e[i] + (bias ? bias[wcol] : 0.f);
            if (act == ACT_RELU) x = fmaxf(x, 0.f);
            else if (act == ACT_MUL) x *= aux[(size_t)grow * ldc + wcol];
            C[(size_t)grow * ldc + wcol] = x;
            if (C16) C16[(size_t)grow * ldc + wcol] = (_Float16)x;
        }
        {
            int wcol = col0 + nc0 + 16;
            float x = acc1.e[i] + (bias ? bias[wcol] : 0.f);
            if (act == ACT_RELU) x = fmaxf(x, 0.f);
            else if (act == ACT_MUL) x *= aux[(size_t)grow * ldc + wcol];
            C[(size_t)grow * ldc + wcol] = x;
            if (C16) C16[(size_t)grow * ldc + wcol] = (_Float16)x;
        }
    }
}

// ---------------- host side -------------------------------------------------
static inline unsigned cdiv256(long long n) { return (unsigned)((n + 255) / 256); }

static inline void launch_gemm(const _Float16* A, int lda, const _Float16* WT, int nout,
                               int K, const float* b, float* C, _Float16* C16, int ldc,
                               const float* aux, int M, int act, hipStream_t s) {
    dim3 grid((M + BM - 1) / BM, nout / BN);
    k_gemm<<<grid, 256, 0, s>>>(A, lda, WT, b, C, C16, ldc, aux, M, K, act);
}

extern "C" void kernel_launch(void* const* d_in, const int* in_sizes, int n_in,
                              void* d_out, int out_size, void* d_ws, size_t ws_size,
                              hipStream_t stream) {
    (void)in_sizes; (void)n_in; (void)out_size; (void)ws_size;

    const float* features = (const float*)d_in[0];
    const float* fused    = (const float*)d_in[1];
    const int*   pool_idx = (const int*)d_in[2];
    const int*   src      = (const int*)d_in[3];
    const int*   dst      = (const int*)d_in[4];
    const float* W0   = (const float*)d_in[5];
    const float* al0  = (const float*)d_in[6];
    const float* ar0  = (const float*)d_in[7];
    const float* b0   = (const float*)d_in[8];
    const float* W1   = (const float*)d_in[9];
    const float* al1  = (const float*)d_in[10];
    const float* ar1  = (const float*)d_in[11];
    const float* b1   = (const float*)d_in[12];
    const float* resW = (const float*)d_in[13];
    const float* Wc1a = (const float*)d_in[14];
    const float* bc1a = (const float*)d_in[15];
    const float* Wc1b = (const float*)d_in[16];
    const float* bc1b = (const float*)d_in[17];
    const float* Wc2a = (const float*)d_in[18];
    const float* bc2a = (const float*)d_in[19];
    const float* Wc2b = (const float*)d_in[20];
    const float* bc2b = (const float*)d_in[21];
    const float* Wa1  = (const float*)d_in[22];
    const float* ba1  = (const float*)d_in[23];
    const float* Wa2  = (const float*)d_in[24];
    const float* ba2  = (const float*)d_in[25];
    float* out = (float*)d_out;

    // ---- workspace carve: f32 region ----
    float* Hb   = (float*)d_ws;                          // N x 128
    float* Fb   = Hb  + (size_t)NN * 128;                // N x 128
    float* Sb   = Fb  + (size_t)NN * 128;                // N x 128
    float* C1b  = Sb  + (size_t)NN * 128;                // N x 256
    float* CONb = C1b + (size_t)NN * 256;                // N x 256
    float* ELOG = CONb + (size_t)NN * 256;               // E x 4
    float* ELb  = ELOG + (size_t)EDGES * 4;              // N x 4
    float* ERb  = ELb  + (size_t)NN * 4;                 // N x 4
    unsigned* EMAXu = (unsigned*)(ERb + (size_t)NN * 4); // N x 4 keys
    float* DENb = (float*)(EMAXu + (size_t)NN * 4);      // N x 4
    // ---- f16 shadow region ----
    _Float16* H16   = (_Float16*)(DENb + (size_t)NN * 4);
    _Float16* S16   = H16  + (size_t)NN * 128;
    _Float16* F16   = S16  + (size_t)NN * 128;
    _Float16* C116  = F16  + (size_t)NN * 128;           // N x 256
    _Float16* CON16 = C116 + (size_t)NN * 256;           // N x 256
    // ---- transposed f16 weights ----
    _Float16* WT0   = CON16 + (size_t)NN * 256;
    _Float16* WT1   = WT0   + 128 * 128;
    _Float16* WTr   = WT1   + 128 * 128;
    _Float16* WTc1a = WTr   + 128 * 128;   // 256 x 128
    _Float16* WTc1b = WTc1a + 256 * 128;   // 128 x 256
    _Float16* WTc2a = WTc1b + 128 * 256;   // 256 x 128
    _Float16* WTc2b = WTc2a + 256 * 128;   // 128 x 256
    _Float16* WTa1  = WTc2b + 128 * 256;   // 256 x 256
    _Float16* WTa2  = WTa1  + 256 * 256;   // 256 x 256

    // 0) weight transpose + convert (tiny)
    k_cvt_wt<<<cdiv256(128 * 128), 256, 0, stream>>>(W0,   WT0,   128, 128);
    k_cvt_wt<<<cdiv256(128 * 128), 256, 0, stream>>>(W1,   WT1,   128, 128);
    k_cvt_wt<<<cdiv256(128 * 128), 256, 0, stream>>>(resW, WTr,   128, 128);
    k_cvt_wt<<<cdiv256(128 * 256), 256, 0, stream>>>(Wc1a, WTc1a, 128, 256);
    k_cvt_wt<<<cdiv256(256 * 128), 256, 0, stream>>>(Wc1b, WTc1b, 256, 128);
    k_cvt_wt<<<cdiv256(128 * 256), 256, 0, stream>>>(Wc2a, WTc2a, 128, 256);
    k_cvt_wt<<<cdiv256(256 * 128), 256, 0, stream>>>(Wc2b, WTc2b, 256, 128);
    k_cvt_wt<<<cdiv256(256 * 256), 256, 0, stream>>>(Wa1,  WTa1,  256, 256);
    k_cvt_wt<<<cdiv256(256 * 256), 256, 0, stream>>>(Wa2,  WTa2,  256, 256);

    // 1) unpool + skip: H = scatter(features) + fused (f32 + f16 shadow)
    k_copy2<<<cdiv256((long long)NN * 128), 256, 0, stream>>>(fused, Hb, H16,
                                                              (long long)NN * 128);
    k_unpool2<<<cdiv256((long long)NPOOL * 128), 256, 0, stream>>>(features, fused,
                                                                   pool_idx, Hb, H16);

    // 2) GAT layer 0
    launch_gemm(H16, 128, WT0, 128, 128, nullptr, Fb, nullptr, 128, nullptr, NN, ACT_LIN, stream);
    k_el_er<<<NN, 128, 0, stream>>>(Fb, al0, ar0, ELb, ERb);
    k_zero<<<cdiv256((long long)NN * 8), 256, 0, stream>>>((float*)EMAXu, (long long)NN * 8);
    k_zero<<<cdiv256((long long)NN * 128), 256, 0, stream>>>(Sb, (long long)NN * 128);
    k_edge1<<<cdiv256((long long)EDGES * 4), 256, 0, stream>>>(src, dst, ELb, ERb, ELOG, EMAXu);
    k_edge2<<<cdiv256((long long)EDGES * 4), 256, 0, stream>>>(dst, ELOG, EMAXu, DENb);
    k_edge3<<<cdiv256((long long)EDGES * 128), 256, 0, stream>>>(src, dst, ELOG, DENb, Fb, Sb, 128);
    k_bias_elu<<<cdiv256((long long)NN * 128), 256, 0, stream>>>(Sb, 128, b0, S16);

    // 3) GAT layer 1 (residual) -> CON[:, 0:128]
    launch_gemm(S16, 128, WT1, 128, 128, nullptr, Fb, nullptr, 128, nullptr, NN, ACT_LIN, stream);
    k_el_er<<<NN, 128, 0, stream>>>(Fb, al1, ar1, ELb, ERb);
    k_zero<<<cdiv256((long long)NN * 8), 256, 0, stream>>>((float*)EMAXu, (long long)NN * 8);
    // residual initializes the aggregation accumulator (non-atomic init, then atomics)
    launch_gemm(S16, 128, WTr, 128, 128, nullptr, CONb, nullptr, 256, nullptr, NN, ACT_LIN, stream);
    k_edge1<<<cdiv256((long long)EDGES * 4), 256, 0, stream>>>(src, dst, ELb, ERb, ELOG, EMAXu);
    k_edge2<<<cdiv256((long long)EDGES * 4), 256, 0, stream>>>(dst, ELOG, EMAXu, DENb);
    k_edge3<<<cdiv256((long long)EDGES * 128), 256, 0, stream>>>(src, dst, ELOG, DENb, Fb, CONb, 256);
    k_bias_elu<<<cdiv256((long long)NN * 128), 256, 0, stream>>>(CONb, 256, b1, CON16);

    // 4) GCT channel branch -> CON[:, 128:256]
    launch_gemm(H16, 128, WTc1a, 256, 128, bc1a, C1b, C116, 256, nullptr, NN, ACT_RELU, stream);
    launch_gemm(C116, 256, WTc1b, 128, 256, bc1b, Fb, nullptr, 128, nullptr, NN, ACT_LIN, stream);
    k_gate<<<NN, 128, 0, stream>>>(Fb, F16);
    launch_gemm(F16, 128, WTc2a, 256, 128, bc2a, C1b, C116, 256, nullptr, NN, ACT_RELU, stream);
    launch_gemm(C116, 256, WTc2b, 128, 256, bc2b, CONb + 128, CON16 + 128, 256, nullptr, NN, ACT_LIN, stream);

    // 5) gated fusion: out = con * (relu(con@Wa1+ba1)@Wa2 + ba2) (fused epilogue)
    launch_gemm(CON16, 256, WTa1, 256, 256, ba1, C1b, C116, 256, nullptr, NN, ACT_RELU, stream);
    launch_gemm(C116, 256, WTa2, 256, 256, ba2, out, nullptr, 256, CONb, NN, ACT_MUL, stream);
}
